// AdapterMamba_74844100100510
// MI455X (gfx1250) — matrix operand order repaired and checked by reading, not compiled, not run on hardware
//
#include <hip/hip_runtime.h>
#include <hip/hip_bf16.h>
#include <math.h>

// ---------------- model constants ----------------
#define LAYERS   4
#define BATCH    4
#define SEQ      2048
#define DM       1024     // d_model
#define DIN      2048     // d_inner
#define NH       64       // heads
#define HP       32       // headdim
#define NS       128      // d_state
#define CONVD    2304     // conv channels
#define DPROJ    4416     // in_proj out dim
#define DCONV    4
#define TOK      (BATCH*SEQ)   // 8192 tokens

typedef __bf16 bf16;
typedef __attribute__((ext_vector_type(16))) __bf16 v16bf;
typedef __attribute__((ext_vector_type(8)))  float  v8f;

// CDNA5 async global->LDS staging (ASYNCcnt path), if the toolchain exposes it.
#if __has_builtin(__builtin_amdgcn_global_load_async_to_lds_b128) && \
    __has_builtin(__builtin_amdgcn_s_wait_asynccnt)
#define USE_ASYNC_LDS 1
typedef int v4i_vec __attribute__((vector_size(4 * sizeof(int))));   // matches builtin params
typedef __attribute__((address_space(3))) v4i_vec* las_v4i;
#else
#define USE_ASYNC_LDS 0
#endif

// ================= weight convert: f32 [K][N] -> bf16 [N][K] (per layer) =================
__global__ void k_convert_t(const float* __restrict__ W, bf16* __restrict__ Wt,
                            int K, int N, long total) {
  long stride = (long)gridDim.x * blockDim.x;
  for (long idx = (long)blockIdx.x * blockDim.x + threadIdx.x; idx < total; idx += stride) {
    long per = (long)K * N;
    int layer = (int)(idx / per);
    long r = idx - (long)layer * per;
    int k = (int)(r / N);
    int n = (int)(r - (long)k * N);
    Wt[(long)layer * per + (long)n * K + k] = (bf16)W[idx];
  }
}

// ================= RMSNorm -> bf16 (D = 1024, 4 elems/thread) =================
__global__ void k_rmsnorm_bf16(const float* __restrict__ in, const float* __restrict__ w,
                               bf16* __restrict__ out) {
  int row = blockIdx.x, tid = threadIdx.x;
  const float* x = in + (size_t)row * DM;
  float v[4]; float ss = 0.f;
#pragma unroll
  for (int i = 0; i < 4; ++i) { v[i] = x[tid + i * 256]; ss += v[i] * v[i]; }
  __shared__ float red[256];
  red[tid] = ss; __syncthreads();
  for (int s = 128; s > 0; s >>= 1) { if (tid < s) red[tid] += red[tid + s]; __syncthreads(); }
  float rstd = rsqrtf(red[0] / (float)DM + 1e-5f);
#pragma unroll
  for (int i = 0; i < 4; ++i) {
    int c = tid + i * 256;
    out[(size_t)row * DM + c] = (bf16)(v[i] * rstd * w[c]);
  }
}

// ================= RMSNorm -> f32 (final) =================
__global__ void k_rmsnorm_f32(const float* __restrict__ in, const float* __restrict__ w,
                              float* __restrict__ out) {
  int row = blockIdx.x, tid = threadIdx.x;
  const float* x = in + (size_t)row * DM;
  float v[4]; float ss = 0.f;
#pragma unroll
  for (int i = 0; i < 4; ++i) { v[i] = x[tid + i * 256]; ss += v[i] * v[i]; }
  __shared__ float red[256];
  red[tid] = ss; __syncthreads();
  for (int s = 128; s > 0; s >>= 1) { if (tid < s) red[tid] += red[tid + s]; __syncthreads(); }
  float rstd = rsqrtf(red[0] / (float)DM + 1e-5f);
#pragma unroll
  for (int i = 0; i < 4; ++i) {
    int c = tid + i * 256;
    out[(size_t)row * DM + c] = v[i] * rstd * w[c];
  }
}

// ================= WMMA GEMM: C[M,N](f32) = A[M,K](bf16) * Bt[N,K](bf16)^T (+resid) ======
// Block: 256 threads (8 waves). Block tile 256(M) x 64(N). K-step 32.
// Wave (wm = wave&3 -> 64 rows, wn = wave>>2 -> 32 cols): 4x2 grid of 16x16 accumulators.
#define LDA 40   // LDS row stride in halfs (80B, multiple of 16B, bank-skewed)

union FragU { uint4 u[2]; v16bf v; };

__global__ __launch_bounds__(256)
void k_gemm_bf16(const bf16* __restrict__ A, const bf16* __restrict__ Bt,
                 float* __restrict__ C, const float* __restrict__ resid,
                 int K, int N) {
  __shared__ __align__(16) bf16 sA[256 * LDA];
  __shared__ __align__(16) bf16 sB[64 * LDA];

  const int tid  = threadIdx.x;
  const int lane = tid & 31;
  const int wave = tid >> 5;
  const int wm   = wave & 3;       // 0..3  -> rows [wm*64, +64)
  const int wn   = wave >> 2;      // 0..1  -> cols [wn*32, +32)
  const int m0   = blockIdx.y * 256;
  const int n0   = blockIdx.x * 64;

  v8f acc[4][2] = {};

  // fragment addressing per CDNA5 ISA layouts
  const int arow = wm * 64 + (lane & 15);
  const int aklo = (lane >> 4) * 8;             // A: lanes16-31 start at K=8
  const int bcol = wn * 32 + (lane & 15);
  const int bkb  = (lane >> 4) * 8;             // B: lanes16-31 start at K=8 (same as A)

  for (int k0 = 0; k0 < K; k0 += 32) {
    // ---- stage A tile 256x32 (1024 16B chunks, 4/thread) + B tile 64x32 (1/thread) ----
#if USE_ASYNC_LDS
#pragma unroll
    for (int i = 0; i < 4; ++i) {
      int chunk = tid + i * 256;
      int row = chunk >> 2, cc = chunk & 3;
      __builtin_amdgcn_global_load_async_to_lds_b128(
          (v4i_vec*)(A + (size_t)(m0 + row) * K + k0 + cc * 8),
          (las_v4i)(&sA[row * LDA + cc * 8]), 0, 0);
    }
    {
      int row = tid >> 2, cc = tid & 3;
      __builtin_amdgcn_global_load_async_to_lds_b128(
          (v4i_vec*)(Bt + (size_t)(n0 + row) * K + k0 + cc * 8),
          (las_v4i)(&sB[row * LDA + cc * 8]), 0, 0);
    }
#else
#pragma unroll
    for (int i = 0; i < 4; ++i) {
      int chunk = tid + i * 256;
      int row = chunk >> 2, cc = chunk & 3;
      const uint4* src = (const uint4*)(A + (size_t)(m0 + row) * K + k0 + cc * 8);
      *(uint4*)(&sA[row * LDA + cc * 8]) = *src;
    }
    {
      int row = tid >> 2, cc = tid & 3;
      const uint4* src = (const uint4*)(Bt + (size_t)(n0 + row) * K + k0 + cc * 8);
      *(uint4*)(&sB[row * LDA + cc * 8]) = *src;
    }
#endif
    if (k0 + 32 < K) {   // global_prefetch_b8 the next K tile
      __builtin_prefetch(A + (size_t)(m0 + tid) * K + k0 + 32, 0, 1);
      __builtin_prefetch(Bt + (size_t)(n0 + (tid >> 2)) * K + k0 + 32, 0, 1);
    }
#if USE_ASYNC_LDS
    __builtin_amdgcn_s_wait_asynccnt(0);
#endif
    __syncthreads();

    FragU fa[4], fb[2];
#pragma unroll
    for (int mi = 0; mi < 4; ++mi) {
      int r = arow + mi * 16;
      fa[mi].u[0] = *(const uint4*)&sA[r * LDA + aklo];
      fa[mi].u[1] = *(const uint4*)&sA[r * LDA + aklo + 16];
    }
#pragma unroll
    for (int ni = 0; ni < 2; ++ni) {
      int c = bcol + ni * 16;
      fb[ni].u[0] = *(const uint4*)&sB[c * LDA + bkb];
      fb[ni].u[1] = *(const uint4*)&sB[c * LDA + bkb + 16];
    }
#pragma unroll
    for (int mi = 0; mi < 4; ++mi)
#pragma unroll
      for (int ni = 0; ni < 2; ++ni)
        acc[mi][ni] = __builtin_amdgcn_wmma_f32_16x16x32_bf16(
            false, fa[mi].v, false, fb[ni].v, (short)0, acc[mi][ni], false, false);
    __syncthreads();
  }

  // ---- epilogue: C/D f32 16x16 layout (VGPR v: M=v+8*(lane>=16), N=lane&15) ----
  const int rbase = m0 + wm * 64 + (lane >> 4) * 8;
  const int cbase = n0 + wn * 32 + (lane & 15);
#pragma unroll
  for (int mi = 0; mi < 4; ++mi)
#pragma unroll
    for (int ni = 0; ni < 2; ++ni)
#pragma unroll
      for (int v = 0; v < 8; ++v) {
        int row = rbase + mi * 16 + v;
        int col = cbase + ni * 16;
        float val = acc[mi][ni][v];
        if (resid) val += resid[(size_t)row * N + col];
        C[(size_t)row * N + col] = val;
      }
}

// ================= conv + silu + dt/dA prep =================
__global__ void k_conv_prep(const float* __restrict__ zx,
                            const float* __restrict__ cw, const float* __restrict__ cb,
                            const float* __restrict__ dtb, const float* __restrict__ alog,
                            float* __restrict__ xbc, float* __restrict__ dtO,
                            float* __restrict__ dAO) {
  int row = blockIdx.x, tid = threadIdx.x;
  int b = row / SEQ, l = row - b * SEQ;
  for (int c = tid; c < CONVD; c += 256) {
    float acc = cb[c];
#pragma unroll
    for (int k = 0; k < DCONV; ++k) {
      int lk = l - (DCONV - 1) + k;
      if (lk >= 0)
        acc += zx[(size_t)(b * SEQ + lk) * DPROJ + DIN + c] * cw[c * DCONV + k];
    }
    xbc[(size_t)row * CONVD + c] = acc / (1.f + expf(-acc));   // silu
  }
  if (tid < NH) {
    float draw = zx[(size_t)row * DPROJ + DIN + CONVD + tid] + dtb[tid];
    float dt = (draw > 20.f) ? draw : log1pf(expf(draw));       // softplus
    float A = -expf(alog[tid]);
    dtO[(size_t)row * NH + tid] = dt;
    dAO[(size_t)row * NH + tid] = expf(dt * A);
  }
}

// ================= sequential SSM scan: one block per (batch, head) =================
// 256 threads: thread owns p = tid>>3 (0..31), n-chunk (tid&7)*16 .. +16  -> 16 state regs.
__global__ __launch_bounds__(256)
void k_scan(const float* __restrict__ xbc, const float* __restrict__ dtB,
            const float* __restrict__ dAB, const float* __restrict__ Dh,
            float* __restrict__ y) {
  const int hh = blockIdx.x, b = blockIdx.y, tid = threadIdx.x;
  const int p = tid >> 3, j = tid & 7, n0 = j * 16;
  float s[16];
#pragma unroll
  for (int i = 0; i < 16; ++i) s[i] = 0.f;
  const float Dv = Dh[hh];

  __shared__ float shB[NS], shC[NS], shX[HP], shS[2];

  for (int l = 0; l < SEQ; ++l) {
    const int row = b * SEQ + l;
    const size_t rb = (size_t)row * CONVD;
    if (tid < 128)      shB[tid]       = xbc[rb + DIN + tid];
    else                shC[tid - 128] = xbc[rb + DIN + NS + (tid - 128)];
    if (tid < HP)       shX[tid]       = xbc[rb + hh * HP + tid];
    if (tid == 0) { shS[0] = dAB[(size_t)row * NH + hh]; shS[1] = dtB[(size_t)row * NH + hh]; }
    __syncthreads();

    const float dAv = shS[0];
    const float dtx = shS[1] * shX[p];
    float cpart = 0.f;
#pragma unroll
    for (int i = 0; i < 16; ++i) {
      s[i] = s[i] * dAv + dtx * shB[n0 + i];
      cpart += s[i] * shC[n0 + i];
    }
    cpart += __shfl_down(cpart, 4, 8);
    cpart += __shfl_down(cpart, 2, 8);
    cpart += __shfl_down(cpart, 1, 8);
    if (j == 0)
      y[(size_t)row * DIN + hh * HP + p] = cpart + Dv * shX[p];
    __syncthreads();
  }
}

// ================= gated RMSNorm -> bf16 (D = 2048, 8 elems/thread) =================
__global__ void k_gated_norm(const float* __restrict__ y, const float* __restrict__ zx,
                             const float* __restrict__ gw, bf16* __restrict__ out) {
  int row = blockIdx.x, tid = threadIdx.x;
  float g[8]; float ss = 0.f;
#pragma unroll
  for (int i = 0; i < 8; ++i) {
    int c = tid + i * 256;
    float z = zx[(size_t)row * DPROJ + c];
    float gg = y[(size_t)row * DIN + c] * (z / (1.f + expf(-z)));
    g[i] = gg; ss += gg * gg;
  }
  __shared__ float red[256];
  red[tid] = ss; __syncthreads();
  for (int s = 128; s > 0; s >>= 1) { if (tid < s) red[tid] += red[tid + s]; __syncthreads(); }
  float rstd = rsqrtf(red[0] / (float)DIN + 1e-5f);
#pragma unroll
  for (int i = 0; i < 8; ++i) {
    int c = tid + i * 256;
    out[(size_t)row * DIN + c] = (bf16)(g[i] * rstd * gw[c]);
  }
}

// ======================= launch =======================
extern "C" void kernel_launch(void* const* d_in, const int* in_sizes, int n_in,
                              void* d_out, int out_size, void* d_ws, size_t ws_size,
                              hipStream_t stream) {
  const float* x    = (const float*)d_in[0];
  const float* Win  = (const float*)d_in[1];
  const float* cw   = (const float*)d_in[2];
  const float* cb   = (const float*)d_in[3];
  const float* dtb  = (const float*)d_in[4];
  const float* alog = (const float*)d_in[5];
  const float* Dp   = (const float*)d_in[6];
  const float* gnw  = (const float*)d_in[7];
  const float* Wout = (const float*)d_in[8];
  const float* bnw  = (const float*)d_in[9];
  const float* fnw  = (const float*)d_in[10];
  float* out = (float*)d_out;

  // workspace carve-up (256B aligned)
  char* ws = (char*)d_ws;
  size_t off = 0;
  auto carve = [&](size_t bytes) { char* p = ws + off; off = (off + bytes + 255) & ~(size_t)255; return p; };
  bf16*  WinT  = (bf16*) carve((size_t)LAYERS * DM * DPROJ * sizeof(bf16));
  bf16*  WoutT = (bf16*) carve((size_t)LAYERS * DIN * DM * sizeof(bf16));
  float* h     = (float*) carve((size_t)TOK * DM * sizeof(float));
  bf16*  hn    = (bf16*)  carve((size_t)TOK * DM * sizeof(bf16));
  float* zx    = (float*) carve((size_t)TOK * DPROJ * sizeof(float));
  float* xbc   = (float*) carve((size_t)TOK * CONVD * sizeof(float));
  float* dtB   = (float*) carve((size_t)TOK * NH * sizeof(float));
  float* dAB   = (float*) carve((size_t)TOK * NH * sizeof(float));
  float* y     = (float*) carve((size_t)TOK * DIN * sizeof(float));
  bf16*  yg    = (bf16*)  carve((size_t)TOK * DIN * sizeof(bf16));
  (void)ws_size; (void)in_sizes; (void)n_in; (void)out_size;

  // residual stream starts as x
  (void)hipMemcpyAsync(h, x, (size_t)TOK * DM * sizeof(float), hipMemcpyDeviceToDevice, stream);

  // one-time per call: bf16 + transpose weights
  {
    long tin = (long)LAYERS * DM * DPROJ;
    k_convert_t<<<2048, 256, 0, stream>>>(Win, WinT, DM, DPROJ, tin);
    long tout = (long)LAYERS * DIN * DM;
    k_convert_t<<<2048, 256, 0, stream>>>(Wout, WoutT, DIN, DM, tout);
  }

  for (int l = 0; l < LAYERS; ++l) {
    const bf16*  wiT = WinT  + (size_t)l * DM * DPROJ;
    const bf16*  woT = WoutT + (size_t)l * DIN * DM;
    const float* cwl = cw + (size_t)l * CONVD * DCONV;
    const float* cbl = cb + (size_t)l * CONVD;
    const float* dbl = dtb + (size_t)l * NH;
    const float* all = alog + (size_t)l * NH;
    const float* ddl = Dp + (size_t)l * NH;
    const float* gwl = gnw + (size_t)l * DIN;
    const float* bwl = bnw + (size_t)l * DM;

    k_rmsnorm_bf16<<<TOK, 256, 0, stream>>>(h, bwl, hn);
    k_gemm_bf16<<<dim3(DPROJ / 64, TOK / 256), 256, 0, stream>>>(hn, wiT, zx, nullptr, DM, DPROJ);
    k_conv_prep<<<TOK, 256, 0, stream>>>(zx, cwl, cbl, dbl, all, xbc, dtB, dAB);
    k_scan<<<dim3(NH, BATCH), 256, 0, stream>>>(xbc, dtB, dAB, ddl, y);
    k_gated_norm<<<TOK, 256, 0, stream>>>(y, zx, gwl, yg);
    k_gemm_bf16<<<dim3(DM / 64, TOK / 256), 256, 0, stream>>>(yg, woT, h, h, DIN, DM);
  }
  k_rmsnorm_f32<<<TOK, 256, 0, stream>>>(h, fnw, out);
}
